// BiRNNModel_51771535786398
// MI455X (gfx1250) — matrix operand order repaired and checked by reading, not compiled, not run on hardware
//
#include <hip/hip_runtime.h>
#include <math.h>

typedef __attribute__((ext_vector_type(16))) _Float16 v16h;
typedef __attribute__((ext_vector_type(8)))  float    v8f;

#define GRU_B 32
#define GRU_S 4096
#define GRU_I 256
#define GRU_H 256
// GEMM: M = B*S = 131072, K = 256, N = 2 dir * 2 layer * 3 gate * 256
// Workspace layout:
//   [0 .. 1572864)          : f16 weights, [dir][layer][gate*256 rows][256 k]
//   [1572864 .. +16384)     : folded biases, [dir][layer][{br,bz,bni,bnh}][256]
#define W16_PER_DIR (2 * 768 * 256)          // 393216 halves per direction
#define BIAS_OFF_BYTES (2 * W16_PER_DIR * 2) // 1572864

// ---------------------------------------------------------------------------
// Prep: convert W_ih (fwd+bwd) to f16, fold biases.
// ---------------------------------------------------------------------------
__global__ void gru_prep(const float* __restrict__ Wf, const float* __restrict__ Wb,
                         const float* __restrict__ bihf, const float* __restrict__ bhhf,
                         const float* __restrict__ bihb, const float* __restrict__ bhhb,
                         _Float16* __restrict__ w16, float* __restrict__ bias) {
  int i = blockIdx.x * blockDim.x + threadIdx.x;
  if (i < W16_PER_DIR) {
    w16[i]               = (_Float16)Wf[i];
    w16[W16_PER_DIR + i] = (_Float16)Wb[i];
  } else {
    int t = i - W16_PER_DIR;
    if (t < 4096) {
      int dir  = t >> 11;
      int l    = (t >> 10) & 1;
      int kind = (t >> 8) & 3;
      int h    = t & 255;
      const float* bih = dir ? bihb : bihf;
      const float* bhh = dir ? bhhb : bhhf;
      int base = l * 768;
      float v;
      if      (kind == 0) v = bih[base + h]       + bhh[base + h];        // r
      else if (kind == 1) v = bih[base + 256 + h] + bhh[base + 256 + h];  // z
      else if (kind == 2) v = bih[base + 512 + h];                        // n (input)
      else                v = bhh[base + 512 + h];                        // n (hidden)
      bias[t] = v;
    }
  }
}

// ---------------------------------------------------------------------------
// WMMA f16 16x16x32 operand layouts (CDNA5 ISA 7.12.2):
//  A (16x32, M x K): lane = M (lanes 0-15 / 16-31 both M=0..15).
//    lanes 0-15 : halves 0..7 = K 0..7,  halves 8..15 = K 16..23
//    lanes 16-31: halves 0..7 = K 8..15, halves 8..15 = K 24..31
//  B (32x16, K x N): lane&15 = N column; this code uses contiguous K per lane-half (A likewise):
//    lanes 0-15 : halves 0..15 = K 0..15
//    lanes 16-31: halves 0..15 = K 16..31
//  C/D (16x16):  VGPR j -> M=j (lanes 0-15) / M=j+8 (lanes 16-31), N = lane&15.
// ---------------------------------------------------------------------------
static __device__ __forceinline__ v16h load_b16(const _Float16* __restrict__ p) {
  v16h v;
  __builtin_memcpy(&v, p, 32);   // 16 contiguous f16 -> two global_load_b128
  return v;
}

static __device__ __forceinline__ v16h cvt_a(const float* __restrict__ p) {
  float4 f0 = *(const float4*)(p);
  float4 f1 = *(const float4*)(p + 4);
  float4 f2 = *(const float4*)(p + 8);
  float4 f3 = *(const float4*)(p + 12);
  v16h v;
  v[0]  = (_Float16)f0.x; v[1]  = (_Float16)f0.y; v[2]  = (_Float16)f0.z; v[3]  = (_Float16)f0.w;
  v[4]  = (_Float16)f1.x; v[5]  = (_Float16)f1.y; v[6]  = (_Float16)f1.z; v[7]  = (_Float16)f1.w;
  v[8]  = (_Float16)f2.x; v[9]  = (_Float16)f2.y; v[10] = (_Float16)f2.z; v[11] = (_Float16)f2.w;
  v[12] = (_Float16)f3.x; v[13] = (_Float16)f3.y; v[14] = (_Float16)f3.z; v[15] = (_Float16)f3.w;
  return v;
}

// Branch-free fast activations (error << f16 GEMM error).
static __device__ __forceinline__ float fast_sigmoid(float x) {
  return __builtin_amdgcn_rcpf(1.0f + __expf(-x));
}
static __device__ __forceinline__ float fast_tanh(float y) {
  float t = __expf(-2.0f * __builtin_fabsf(y));
  float r = (1.0f - t) * __builtin_amdgcn_rcpf(1.0f + t);
  return __builtin_copysignf(r, y);
}

// ---------------------------------------------------------------------------
// Main: each wave computes a 16-row strip against all 3072 gate columns.
// ---------------------------------------------------------------------------
__global__ __launch_bounds__(256) void gru_main(const float* __restrict__ x,
                                                const _Float16* __restrict__ w16,
                                                const float* __restrict__ bias,
                                                float* __restrict__ out) {
  const int lane   = threadIdx.x & 31;
  const int wave   = threadIdx.x >> 5;
  const int lane16 = lane & 15;
  const bool hi    = lane >= 16;
  const int m0     = blockIdx.x * 128 + wave * 16;     // strip base row (M)
  const int kba    = hi ? 16 : 0;                      // A operand K base (contiguous)
  const int kbb    = hi ? 16 : 0;                      // B operand K base (contiguous)

  // Load A strip: row (m0+lane16), all K=256, f32 -> f16 in WMMA A layout.
  const float* rowptr = x + (size_t)(m0 + lane16) * GRU_I;
  v16h a[8];
#pragma unroll
  for (int c = 0; c < 8; ++c) a[c] = cvt_a(rowptr + 32 * c + kba);

  for (int dir = 0; dir < 2; ++dir) {
    const _Float16* wd = w16 + dir * W16_PER_DIR;
    const float*    bd = bias + dir * 2048;
    for (int l = 0; l < 2; ++l) {
      const _Float16* wl = wd + l * (768 * 256);
      const float*    bl = bd + l * 1024;

      // Output element offsets (fit in 32 bits: 2^27 total elements).
      unsigned ob[8];
#pragma unroll
      for (int j = 0; j < 8; ++j) {
        const int m = m0 + j + (hi ? 8 : 0);           // output row (M) for acc[j]
        const int b = m >> 12;                         // / S
        const int s = m & (GRU_S - 1);
        const int row = (dir == 0) ? (s * 2 + l)
                                   : (2 * GRU_S + ((GRU_S - s) & (GRU_S - 1)) * 2 + l);
        ob[j] = ((unsigned)b * (4 * GRU_S) + (unsigned)row) * GRU_H;
      }

      for (int ht = 0; ht < 16; ++ht) {
        const int g = ht * 16 + lane16;                // this lane's output column
        const _Float16* wr = wl + (size_t)g * 256;     // gate-r row for column g
        v8f accR = {}, accZ = {}, accN = {};
#pragma unroll
        for (int c = 0; c < 8; ++c) {
          const int k0 = 32 * c + kbb;
          v16h bR = load_b16(wr + k0);
          v16h bZ = load_b16(wr + 65536 + k0);
          v16h bN = load_b16(wr + 131072 + k0);
          accR = __builtin_amdgcn_wmma_f32_16x16x32_f16(false, a[c], false, bR, (short)0, accR, false, false);
          accZ = __builtin_amdgcn_wmma_f32_16x16x32_f16(false, a[c], false, bZ, (short)0, accZ, false, false);
          accN = __builtin_amdgcn_wmma_f32_16x16x32_f16(false, a[c], false, bN, (short)0, accN, false, false);
        }
        const float brv  = bl[g];
        const float bzv  = bl[256 + g];
        const float bniv = bl[512 + g];
        const float bnhv = bl[768 + g];
#pragma unroll
        for (int j = 0; j < 8; ++j) {
          const float r = fast_sigmoid(accR[j] + brv);
          const float z = fast_sigmoid(accZ[j] + bzv);
          const float n = fast_tanh(accN[j] + bniv + r * bnhv);
          out[ob[j] + (unsigned)g] = (1.0f - z) * n;
        }
      }
    }
  }
}

// ---------------------------------------------------------------------------
extern "C" void kernel_launch(void* const* d_in, const int* in_sizes, int n_in,
                              void* d_out, int out_size, void* d_ws, size_t ws_size,
                              hipStream_t stream) {
  const float* x    = (const float*)d_in[0];
  const float* Wf   = (const float*)d_in[1];
  // d_in[2] = W_hh_fwd (unused: h == 0)
  const float* bihf = (const float*)d_in[3];
  const float* bhhf = (const float*)d_in[4];
  const float* Wb   = (const float*)d_in[5];
  // d_in[6] = W_hh_bwd (unused)
  const float* bihb = (const float*)d_in[7];
  const float* bhhb = (const float*)d_in[8];

  _Float16* w16 = (_Float16*)d_ws;
  float* bias   = (float*)((char*)d_ws + BIAS_OFF_BYTES);
  float* out    = (float*)d_out;

  // prep: 393216 weight threads + 4096 bias threads
  const int prep_threads = W16_PER_DIR + 4096;
  gru_prep<<<(prep_threads + 255) / 256, 256, 0, stream>>>(Wf, Wb, bihf, bhhf, bihb, bhhb, w16, bias);

  // main: M=131072 rows / (8 waves * 16 rows) = 1024 workgroups
  gru_main<<<1024, 256, 0, stream>>>(x, w16, bias, out);
}
